// STACAME_minibatch_77644418777395
// MI455X (gfx1250) — compile-verified, matrix operand order checked by reading, not run on hardware
//
#include <hip/hip_runtime.h>
#include <hip/hip_bf16.h>
#include <math.h>

#define NNODES 50000
#define NEDGES 400000
#define INDIM  1024
#define OUTDIM 256

typedef __attribute__((ext_vector_type(16))) __bf16 v16bf;
typedef __attribute__((ext_vector_type(8)))  float  v8f;

union FragU { uint4 u[2]; v16bf v; };

// scalar fp32 -> bf16, RNE (used in small weight-transpose kernels)
__device__ __forceinline__ unsigned short f2bf(float f) {
  unsigned u = __float_as_uint(f);
  u += 0x7fffu + ((u >> 16) & 1u);
  return (unsigned short)(u >> 16);
}
// fast packed fp32x2 -> bf16x2: rounding-bias add + single v_perm_b32 byte pack
__device__ __forceinline__ unsigned pk2(float a, float b) {
  unsigned ua = __float_as_uint(a) + 0x8000u;
  unsigned ub = __float_as_uint(b) + 0x8000u;
  return __builtin_amdgcn_perm(ub, ua, 0x07060302u);
}
// monotone float -> uint key (segment-max via u32 atomicMax)
__device__ __forceinline__ unsigned fkey(float f) {
  unsigned b = __float_as_uint(f);
  return (b & 0x80000000u) ? ~b : (b | 0x80000000u);
}
__device__ __forceinline__ float funkey(unsigned k) {
  unsigned b = (k & 0x80000000u) ? (k & 0x7fffffffu) : ~k;
  return __uint_as_float(b);
}

__device__ __forceinline__ v16bf load_bf_frag(const unsigned short* p) {
  FragU f;
  f.u[0] = *(const uint4*)(p + 0);
  f.u[1] = *(const uint4*)(p + 16);
  return f.v;
}
__device__ __forceinline__ v16bf load_f32_frag(const float* ar) {
  float4 f0 = *(const float4*)(ar + 0);
  float4 f1 = *(const float4*)(ar + 4);
  float4 f2 = *(const float4*)(ar + 16);
  float4 f3 = *(const float4*)(ar + 20);
  FragU f;
  f.u[0] = make_uint4(pk2(f0.x,f0.y), pk2(f0.z,f0.w), pk2(f1.x,f1.y), pk2(f1.z,f1.w));
  f.u[1] = make_uint4(pk2(f2.x,f2.y), pk2(f2.z,f2.w), pk2(f3.x,f3.y), pk2(f3.z,f3.w));
  return f.v;
}

// ---------------------------------------------------------------------------
// Zero-init: h1 output region (atomic accumulation target), denom, max-keys
// ---------------------------------------------------------------------------
__global__ void init_kernel(float* __restrict__ out1, float* __restrict__ denom,
                            unsigned* __restrict__ mkey, int n_out1, int n_nodes) {
  int i = blockIdx.x * blockDim.x + threadIdx.x;
  if (i < n_out1) out1[i] = 0.0f;
  if (i < n_nodes) { denom[i] = 0.0f; mkey[i] = 0u; }
}

// W1 [1024x256] f32 -> W1^T [256x1024] bf16
__global__ void cvtW1_kernel(const float* __restrict__ W1, unsigned short* __restrict__ w1T) {
  int idx = blockIdx.x * blockDim.x + threadIdx.x;
  if (idx >= OUTDIM * INDIM) return;
  int n = idx / INDIM, k = idx - n * INDIM;
  w1T[idx] = f2bf(W1[(size_t)k * OUTDIM + n]);
}
// W4 [256x1024] f32 -> W4^T [1024x256] bf16
__global__ void cvtW4_kernel(const float* __restrict__ W4, unsigned short* __restrict__ w4T) {
  int idx = blockIdx.x * blockDim.x + threadIdx.x;
  if (idx >= INDIM * OUTDIM) return;
  int n = idx / OUTDIM, k = idx - n * OUTDIM;
  w4T[idx] = f2bf(W4[(size_t)k * INDIM + n]);
}

// ---------------------------------------------------------------------------
// bf16 WMMA GEMM: C[M,NCOLS] = A[M,K] * B[K,NCOLS], B given transposed (bf16).
// One wave -> 32x64 output tile: 2 A-frags x 4 B-frags -> 8 WMMAs per k-step;
// each B-frag reused by both M-halves. K / NCOLS / M compile-time so address
// math is shifts + immediate offsets. Tail tile (M%32==16): A-row clamp for
// the upper half, upper-half stores skipped under one wave-uniform branch.
// ---------------------------------------------------------------------------
template <bool AF32, int K, int NCOLS>
__global__ void __launch_bounds__(256)
wmma_gemm_kernel(const void* __restrict__ Ap, const unsigned short* __restrict__ BT,
                 float* __restrict__ C, int njobs) {
  const int wid = (int)((blockIdx.x * blockDim.x + threadIdx.x) >> 5);
  if (wid >= njobs) return;                 // uniform per wave: EXEC all-1 inside
  const int lane = threadIdx.x & 31;
  constexpr int NT64 = NCOLS / 64;
  const int mt2 = wid / NT64;
  const int n64 = (wid - mt2 * NT64) << 6;

  const int rlane   = lane & 15;
  const int halfsel = (lane >> 4) << 3;     // lane<16 -> K+0, lane>=16 -> K+8
  const int r0   = mt2 * 32 + rlane;        // always < NNODES
  const bool full = (mt2 * 32 + 32) <= NNODES;   // wave-uniform
  const int r1   = full ? (r0 + 16) : (NNODES - 1);  // tail: duplicate read, no store

  v8f zero = {0.f,0.f,0.f,0.f,0.f,0.f,0.f,0.f};
  v8f acc[2][4];
#pragma unroll
  for (int m = 0; m < 2; ++m)
#pragma unroll
    for (int n = 0; n < 4; ++n) acc[m][n] = zero;

  const float*          Af = (const float*)Ap;
  const unsigned short* Ab = (const unsigned short*)Ap;

  for (int k0 = 0; k0 < K; k0 += 32) {
    const int kb = k0 + halfsel;
    v16bf a0, a1;
    if (AF32) {
      a0 = load_f32_frag(Af + (size_t)r0 * K + kb);
      a1 = load_f32_frag(Af + (size_t)r1 * K + kb);
    } else {
      a0 = load_bf_frag(Ab + (size_t)r0 * K + kb);
      a1 = load_bf_frag(Ab + (size_t)r1 * K + kb);
    }
#pragma unroll
    for (int nt = 0; nt < 4; ++nt) {
      const v16bf b = load_bf_frag(BT + (size_t)(n64 + nt * 16 + rlane) * K + kb);
      acc[0][nt] = __builtin_amdgcn_wmma_f32_16x16x32_bf16(
          false, a0, false, b, (short)0, acc[0][nt], false, false);
      acc[1][nt] = __builtin_amdgcn_wmma_f32_16x16x32_bf16(
          false, a1, false, b, (short)0, acc[1][nt], false, false);
    }
  }

  // C/D layout: VGPR j -> row j (lanes 0-15) / row j+8 (lanes 16-31), col = lane&15
  const int rb0 = mt2 * 32 + halfsel;
#pragma unroll
  for (int nt = 0; nt < 4; ++nt) {
    float* cp = C + (size_t)rb0 * NCOLS + (n64 + nt * 16 + rlane);
#pragma unroll
    for (int j = 0; j < 8; ++j)
      cp[(size_t)j * NCOLS] = acc[0][nt][j];
  }
  if (full) {                                // single wave-uniform branch
#pragma unroll
    for (int nt = 0; nt < 4; ++nt) {
      float* cp = C + (size_t)(rb0 + 16) * NCOLS + (n64 + nt * 16 + rlane);
#pragma unroll
      for (int j = 0; j < 8; ++j)
        cp[(size_t)j * NCOLS] = acc[1][nt][j];
    }
  }
}

// ---------------------------------------------------------------------------
// Per-node attention logits: alpha_s = h . att_src ; alpha_d = h . att_dst
// ---------------------------------------------------------------------------
__global__ void alpha_kernel(const float* __restrict__ h, const float* __restrict__ asv,
                             const float* __restrict__ adv, float* __restrict__ as_out,
                             float* __restrict__ ad_out) {
  int wid  = (int)((blockIdx.x * blockDim.x + threadIdx.x) >> 5);
  int lane = threadIdx.x & 31;
  if (wid >= NNODES) return;
  const float* hr = h + (size_t)wid * OUTDIM;
  float ss = 0.f, sd = 0.f;
#pragma unroll
  for (int j = 0; j < 8; ++j) {
    int c = lane + 32 * j;
    float hv = hr[c];
    ss += hv * asv[c];
    sd += hv * adv[c];
  }
#pragma unroll
  for (int off = 16; off > 0; off >>= 1) {
    ss += __shfl_xor(ss, off, 32);
    sd += __shfl_xor(sd, off, 32);
  }
  if (lane == 0) { as_out[wid] = ss; ad_out[wid] = sd; }
}

// edge logits + leaky_relu + segment-max (monotone-key atomicMax)
__global__ void edge_logit_kernel(const int* __restrict__ src, const int* __restrict__ dst,
                                  const float* __restrict__ as, const float* __restrict__ ad,
                                  float* __restrict__ e, unsigned* __restrict__ mkey) {
  int i = blockIdx.x * blockDim.x + threadIdx.x;
  if (i >= NEDGES) return;
  int d = dst[i];
  float v = as[src[i]] + ad[d];
  v = (v >= 0.f) ? v : 0.2f * v;
  e[i] = v;
  atomicMax(&mkey[d], fkey(v));
}

// decode segment max; nodes with no in-edges (key==0 sentinel) -> 0
__global__ void node_m_kernel(const unsigned* __restrict__ mkey, float* __restrict__ m) {
  int i = blockIdx.x * blockDim.x + threadIdx.x;
  if (i >= NNODES) return;
  unsigned k = mkey[i];
  float v = 0.f;
  if (k != 0u) {
    float f = funkey(k);
    if (fabsf(f) < __builtin_inff()) v = f;
  }
  m[i] = v;
}

// ex = exp(e - m[dst]); denom[dst] += ex
__global__ void edge_exp_kernel(const int* __restrict__ dst, const float* __restrict__ e,
                                const float* __restrict__ m, float* __restrict__ ex,
                                float* __restrict__ denom) {
  int i = blockIdx.x * blockDim.x + threadIdx.x;
  if (i >= NEDGES) return;
  int d = dst[i];
  float v = expf(e[i] - m[d]);
  ex[i] = v;
  atomicAdd(&denom[d], v);
}

// one wave per edge: out1[dst,:] += (ex/denom[dst]) * h[src,:]
// b128 gathers of h[src], coalesced scalar float atomics into out1[dst]
__global__ void edge_agg_kernel(const int* __restrict__ src, const int* __restrict__ dst,
                                const float* __restrict__ ex, const float* __restrict__ denom,
                                const float* __restrict__ h, float* __restrict__ out1) {
  int eid  = (int)((blockIdx.x * blockDim.x + threadIdx.x) >> 5);
  int lane = threadIdx.x & 31;
  if (eid >= NEDGES) return;
  int s = src[eid], d = dst[eid];
  float w = ex[eid] / (denom[d] + 1e-16f);
  const float* hs = h + (size_t)s * OUTDIM;
  float* od = out1 + (size_t)d * OUTDIM;
#pragma unroll
  for (int p = 0; p < 2; ++p) {
    int c = 4 * lane + 128 * p;
    float4 hv = *(const float4*)(hs + c);
    atomicAdd(&od[c + 0], w * hv.x);
    atomicAdd(&od[c + 1], w * hv.y);
    atomicAdd(&od[c + 2], w * hv.z);
    atomicAdd(&od[c + 3], w * hv.w);
  }
}

// in-place ELU on h1 output + bf16 copy for GEMM2
__global__ void elu_kernel(float* __restrict__ out1, unsigned short* __restrict__ h1B, int n) {
  int i = blockIdx.x * blockDim.x + threadIdx.x;
  if (i >= n) return;
  float x = out1[i];
  float y = (x > 0.f) ? x : (expf(x) - 1.0f);
  out1[i] = y;
  h1B[i] = f2bf(y);
}

// ---------------------------------------------------------------------------
extern "C" void kernel_launch(void* const* d_in, const int* in_sizes, int n_in,
                              void* d_out, int out_size, void* d_ws, size_t ws_size,
                              hipStream_t stream) {
  const float* features = (const float*)d_in[0];
  const int*   ei       = (const int*)d_in[1];
  const int*   src      = ei;
  const int*   dst      = ei + NEDGES;
  const float* W1       = (const float*)d_in[2];
  const float* att_src  = (const float*)d_in[3];
  const float* att_dst  = (const float*)d_in[4];
  const float* W4       = (const float*)d_in[5];

  float* out_h1 = (float*)d_out;
  float* out_h4 = out_h1 + (size_t)NNODES * OUTDIM;

  char* wsb = (char*)d_ws;
  size_t off = 0;
  auto carve = [&](size_t bytes) -> void* {
    off = (off + 255) & ~(size_t)255;
    void* p = wsb + off;
    off += bytes;
    return p;
  };
  float*          h     = (float*)carve((size_t)NNODES * OUTDIM * 4);
  float*          as    = (float*)carve((size_t)NNODES * 4);
  float*          ad    = (float*)carve((size_t)NNODES * 4);
  unsigned*       mkey  = (unsigned*)carve((size_t)NNODES * 4);
  float*          m     = (float*)carve((size_t)NNODES * 4);
  float*          denom = (float*)carve((size_t)NNODES * 4);
  float*          e     = (float*)carve((size_t)NEDGES * 4);
  float*          ex    = (float*)carve((size_t)NEDGES * 4);
  unsigned short* w1T   = (unsigned short*)carve((size_t)OUTDIM * INDIM * 2);
  unsigned short* w4T   = (unsigned short*)carve((size_t)INDIM * OUTDIM * 2);
  unsigned short* h1B   = (unsigned short*)carve((size_t)NNODES * OUTDIM * 2);
  (void)ws_size; (void)in_sizes; (void)n_in; (void)out_size;

  const int TB = 256;
  const int n_out1 = NNODES * OUTDIM;                 // 12.8M

  init_kernel<<<(n_out1 + TB - 1) / TB, TB, 0, stream>>>(out_h1, denom, mkey, n_out1, NNODES);

  const int wn = OUTDIM * INDIM;                      // 262144
  cvtW1_kernel<<<(wn + TB - 1) / TB, TB, 0, stream>>>(W1, w1T);
  cvtW4_kernel<<<(wn + TB - 1) / TB, TB, 0, stream>>>(W4, w4T);

  const int MT2 = (NNODES + 31) / 32;                 // 1563 (incl. 16-row tail)

  // GEMM1: h = features(fp32->bf16) @ W1   [50000x1024]x[1024x256]
  const int njobs1 = MT2 * (OUTDIM / 64);             // 6252 waves
  wmma_gemm_kernel<true, INDIM, OUTDIM><<<(njobs1 * 32 + TB - 1) / TB, TB, 0, stream>>>(
      features, w1T, h, njobs1);

  alpha_kernel<<<(NNODES * 32 + TB - 1) / TB, TB, 0, stream>>>(h, att_src, att_dst, as, ad);

  edge_logit_kernel<<<(NEDGES + TB - 1) / TB, TB, 0, stream>>>(src, dst, as, ad, e, mkey);
  node_m_kernel<<<(NNODES + TB - 1) / TB, TB, 0, stream>>>(mkey, m);
  edge_exp_kernel<<<(NEDGES + TB - 1) / TB, TB, 0, stream>>>(dst, e, m, ex, denom);
  edge_agg_kernel<<<(NEDGES * 32 + TB - 1) / TB, TB, 0, stream>>>(src, dst, ex, denom, h, out_h1);

  elu_kernel<<<(n_out1 + TB - 1) / TB, TB, 0, stream>>>(out_h1, h1B, n_out1);

  // GEMM2: h4 = h1(bf16) @ W4   [50000x256]x[256x1024]
  const int njobs2 = MT2 * (INDIM / 64);              // 25008 waves
  wmma_gemm_kernel<false, OUTDIM, INDIM><<<(njobs2 * 32 + TB - 1) / TB, TB, 0, stream>>>(
      h1B, w4T, out_h4, njobs2);
}
